// OctreeConvBnRelu_8624294331029
// MI455X (gfx1250) — matrix operand order repaired and checked by reading, not compiled, not run on hardware
//
#include <hip/hip_runtime.h>
#include <hip/hip_bf16.h>

typedef __attribute__((ext_vector_type(16))) _Float16 v16h;
typedef __attribute__((ext_vector_type(8)))  _Float16 v8h;
typedef __attribute__((ext_vector_type(8)))  float    v8f;

#define N_NODES 150000
#define C_IN    32
#define C_OUT   64
#define KNBR    27
#define N_TILES (N_NODES / 16)          // 9375 exactly
#define WT_ELEMS (KNBR * C_OUT * C_IN)  // 55296 halves = 110,592 B
#define W_HALF  ((N_TILES + 1) / 2)     // 4688 wave-pairs

// ---------------------------------------------------------------------------
// Pack x (f32, [32][150000]) -> x_t (f16, [150001][32]); row 150000 = zeros.
// ---------------------------------------------------------------------------
__global__ __launch_bounds__(256) void pack_x_kernel(const float* __restrict__ x,
                                                     _Float16* __restrict__ xt) {
    int node = blockIdx.x * 256 + threadIdx.x;
    if (node > N_NODES) return;
    v8h h[4];
    if (node < N_NODES) {
        #pragma unroll
        for (int c = 0; c < 32; ++c)
            h[c >> 3][c & 7] = (_Float16)x[c * N_NODES + node];
    } else {
        #pragma unroll
        for (int q = 0; q < 4; ++q) h[q] = (v8h)0;
    }
    v8h* dst = (v8h*)(xt + node * 32);
    #pragma unroll
    for (int q = 0; q < 4; ++q) dst[q] = h[q];
}

// ---------------------------------------------------------------------------
// Pack W (f32, [64][32][27]) -> Wt (f16, [27][64][32])  (neighbor-major K)
// ---------------------------------------------------------------------------
__global__ __launch_bounds__(256) void pack_w_kernel(const float* __restrict__ W,
                                                     _Float16* __restrict__ wt) {
    int i = blockIdx.x * 256 + threadIdx.x;
    if (i >= WT_ELEMS) return;
    int s = i >> 11;           // / (64*32)
    int o = (i >> 5) & 63;
    int k = i & 31;            // input channel
    wt[i] = (_Float16)W[(o * 32 + k) * 27 + s];
}

// ---------------------------------------------------------------------------
// Gather-GEMM: y[64][150000] = Wt x gathered(x_t).
// Wt staged once per block into 110KB of LDS (320KB/WGP on CDNA5).
// Each wave: 2 node tiles (32 nodes) x all 64 out channels -> 8 accumulators.
// Per K-slice: 2 neigh loads + 4 b128 gather loads (vmem) + 8 ds_load_b128 (A)
//              + 8x V_WMMA_F32_16X16X32_F16.
//
// f16 A fragment: lanes 0-15: M=lane, K=0..15;
//   lanes 16-31: M=lane-16, K=16..31  -> two 16B LDS loads at k0=hi*16.
// f16 B fragment: lanes 0-15: N=lane, K=0..15; lanes 16-31: K=16..31
//   -> 32 contiguous bytes of x_t[gathered node] at half-offset hi*16.
// ---------------------------------------------------------------------------
__global__ __launch_bounds__(256) void conv_wmma_kernel(
    const _Float16* __restrict__ xt,    // [150001][32]
    const _Float16* __restrict__ wt,    // [27][64][32]
    const int*      __restrict__ neigh, // [150000][27]
    float*          __restrict__ y)     // [64][150000]
{
    __shared__ _Float16 sW[WT_ELEMS];   // 110,592 B of the 320KB WGP LDS

    // cooperative fill: 6912 x 16B chunks, 27 per thread
    {
        const v8h* src = (const v8h*)wt;
        v8h*       dst = (v8h*)sW;
        for (int i = threadIdx.x; i < WT_ELEMS / 8; i += 256)
            dst[i] = src[i];
    }
    __syncthreads();

    const int lane = threadIdx.x & 31;
    const int wave = threadIdx.x >> 5;
    const int w    = blockIdx.x * 8 + wave;   // 0..4687 (exact: 586 blocks)
    const int nn   = lane & 15;   // node-in-tile (B/N) and out-channel row (A/M)
    const int hi   = lane >> 4;   // half-wave selector

    const int  tile0 = w;
    int        tile1 = w + W_HALF;
    const bool has2  = (tile1 < N_TILES);
    if (!has2) tile1 = tile0;     // alias: keeps EXEC full for WMMA, store skipped

    const int* nrow0 = neigh + (tile0 * 16 + nn) * KNBR;
    const int* nrow1 = neigh + (tile1 * 16 + nn) * KNBR;

    v8f acc00 = {}, acc01 = {}, acc02 = {}, acc03 = {};
    v8f acc10 = {}, acc11 = {}, acc12 = {}, acc13 = {};

    for (int s = 0; s < KNBR; ++s) {
        // ---- B: gather one neighbor's 32 channels (64B contiguous) x 2 tiles
        const int g0 = nrow0[s];
        const int g1 = nrow1[s];
        const v8h* bp0 = (const v8h*)(xt + (g0 << 5) + (hi << 4));
        const v8h* bp1 = (const v8h*)(xt + (g1 << 5) + (hi << 4));
        v8h b00 = bp0[0], b01 = bp0[1];
        v8h b10 = bp1[0], b11 = bp1[1];
        v16h B0 = __builtin_shufflevector(b00, b01, 0,1,2,3,4,5,6,7,8,9,10,11,12,13,14,15);
        v16h B1 = __builtin_shufflevector(b10, b11, 0,1,2,3,4,5,6,7,8,9,10,11,12,13,14,15);

        // ---- A: 4 output-channel tiles of Wt[s] from LDS ----
        const _Float16* wb = sW + (s << 11) + (nn << 5) + (hi << 4);
        v8h a00 = *(const v8h*)(wb);         v8h a01 = *(const v8h*)(wb + 8);
        v8h a10 = *(const v8h*)(wb + 512);   v8h a11 = *(const v8h*)(wb + 520);
        v8h a20 = *(const v8h*)(wb + 1024);  v8h a21 = *(const v8h*)(wb + 1032);
        v8h a30 = *(const v8h*)(wb + 1536);  v8h a31 = *(const v8h*)(wb + 1544);
        v16h A0 = __builtin_shufflevector(a00, a01, 0,1,2,3,4,5,6,7,8,9,10,11,12,13,14,15);
        v16h A1 = __builtin_shufflevector(a10, a11, 0,1,2,3,4,5,6,7,8,9,10,11,12,13,14,15);
        v16h A2 = __builtin_shufflevector(a20, a21, 0,1,2,3,4,5,6,7,8,9,10,11,12,13,14,15);
        v16h A3 = __builtin_shufflevector(a30, a31, 0,1,2,3,4,5,6,7,8,9,10,11,12,13,14,15);

        // ---- 8 WMMAs: A reused across both node tiles ----
        acc00 = __builtin_amdgcn_wmma_f32_16x16x32_f16(false, A0, false, B0, (short)0, acc00, false, false);
        acc10 = __builtin_amdgcn_wmma_f32_16x16x32_f16(false, A0, false, B1, (short)0, acc10, false, false);
        acc01 = __builtin_amdgcn_wmma_f32_16x16x32_f16(false, A1, false, B0, (short)0, acc01, false, false);
        acc11 = __builtin_amdgcn_wmma_f32_16x16x32_f16(false, A1, false, B1, (short)0, acc11, false, false);
        acc02 = __builtin_amdgcn_wmma_f32_16x16x32_f16(false, A2, false, B0, (short)0, acc02, false, false);
        acc12 = __builtin_amdgcn_wmma_f32_16x16x32_f16(false, A2, false, B1, (short)0, acc12, false, false);
        acc03 = __builtin_amdgcn_wmma_f32_16x16x32_f16(false, A3, false, B0, (short)0, acc03, false, false);
        acc13 = __builtin_amdgcn_wmma_f32_16x16x32_f16(false, A3, false, B1, (short)0, acc13, false, false);
    }

    // ---- store D: lane L, VGPR r -> channel (t*16 + hi*8 + r), node col ----
    {
        float* yb = y + tile0 * 16 + nn;
        #pragma unroll
        for (int r = 0; r < 8; ++r) {
            int m = hi * 8 + r;
            yb[(0 * 16 + m) * N_NODES] = acc00[r];
            yb[(1 * 16 + m) * N_NODES] = acc01[r];
            yb[(2 * 16 + m) * N_NODES] = acc02[r];
            yb[(3 * 16 + m) * N_NODES] = acc03[r];
        }
    }
    if (has2) {
        float* yb = y + tile1 * 16 + nn;
        #pragma unroll
        for (int r = 0; r < 8; ++r) {
            int m = hi * 8 + r;
            yb[(0 * 16 + m) * N_NODES] = acc10[r];
            yb[(1 * 16 + m) * N_NODES] = acc11[r];
            yb[(2 * 16 + m) * N_NODES] = acc12[r];
            yb[(3 * 16 + m) * N_NODES] = acc13[r];
        }
    }
}

// ---------------------------------------------------------------------------
// Deterministic per-channel sum / sumsq: stage 1 (64 x 8 blocks)
// ---------------------------------------------------------------------------
#define CHUNK (N_NODES / 8)   // 18750
__global__ __launch_bounds__(256) void reduce_partial_kernel(const float* __restrict__ y,
                                                             float* __restrict__ partial) {
    const int ch = blockIdx.x, part = blockIdx.y;
    const float* p = y + ch * N_NODES + part * CHUNK;
    float s = 0.f, s2 = 0.f;
    for (int i = threadIdx.x; i < CHUNK; i += 256) {
        float v = p[i];
        s += v; s2 += v * v;
    }
    __shared__ float sh[512];
    sh[threadIdx.x] = s; sh[256 + threadIdx.x] = s2;
    __syncthreads();
    for (int off = 128; off > 0; off >>= 1) {
        if (threadIdx.x < off) {
            sh[threadIdx.x]       += sh[threadIdx.x + off];
            sh[256 + threadIdx.x] += sh[256 + threadIdx.x + off];
        }
        __syncthreads();
    }
    if (threadIdx.x == 0) {
        partial[ch * 8 + part]       = sh[0];
        partial[512 + ch * 8 + part] = sh[256];
    }
}

// stage 2: 64 threads -> mean and rsqrt(var + eps)
__global__ void finalize_stats_kernel(const float* __restrict__ partial,
                                      float* __restrict__ stats) {
    int ch = threadIdx.x;
    float s = 0.f, s2 = 0.f;
    #pragma unroll
    for (int p = 0; p < 8; ++p) {
        s  += partial[ch * 8 + p];
        s2 += partial[512 + ch * 8 + p];
    }
    const float inv_n = 1.0f / (float)N_NODES;
    float mean = s * inv_n;
    float var  = s2 * inv_n - mean * mean;
    stats[ch]      = mean;
    stats[64 + ch] = rsqrtf(var + 1e-3f);
}

// ---------------------------------------------------------------------------
// BN + ReLU: out = max(0, (y-mean)*invstd*gamma + beta); grid.y = channel
// ---------------------------------------------------------------------------
__global__ __launch_bounds__(256) void bn_relu_kernel(const float* __restrict__ y,
                                                      const float* __restrict__ stats,
                                                      const float* __restrict__ gamma,
                                                      const float* __restrict__ beta,
                                                      float* __restrict__ out) {
    const int ch = blockIdx.y;
    const int n  = blockIdx.x * 256 + threadIdx.x;
    if (n >= N_NODES) return;
    const int i = ch * N_NODES + n;
    float v = (y[i] - stats[ch]) * stats[64 + ch] * gamma[ch] + beta[ch];
    out[i] = v > 0.f ? v : 0.f;
}

// ---------------------------------------------------------------------------
extern "C" void kernel_launch(void* const* d_in, const int* in_sizes, int n_in,
                              void* d_out, int out_size, void* d_ws, size_t ws_size,
                              hipStream_t stream) {
    const float* x     = (const float*)d_in[0];   // (1,32,150000,1)
    const int*   neigh = (const int*)  d_in[1];   // (150000,27)
    const float* W     = (const float*)d_in[2];   // (64,32,27)
    const float* gamma = (const float*)d_in[3];
    const float* beta  = (const float*)d_in[4];
    float* out = (float*)d_out;

    // workspace layout (bytes, 256-aligned)
    char* ws = (char*)d_ws;
    _Float16* xt      = (_Float16*)(ws);                 // 150001*32*2 = 9,600,064
    _Float16* wt      = (_Float16*)(ws + 9600256);       // 110,592
    float*    y       = (float*)   (ws + 9710848);       // 64*150000*4 = 38,400,000
    float*    partial = (float*)   (ws + 48110848);      // 1024 floats
    float*    stats   = (float*)   (ws + 48115200);      // 128 floats

    pack_x_kernel<<<(N_NODES + 256) / 256, 256, 0, stream>>>(x, xt);
    pack_w_kernel<<<(WT_ELEMS + 255) / 256, 256, 0, stream>>>(W, wt);

    conv_wmma_kernel<<<W_HALF / 8, 256, 0, stream>>>(xt, wt, neigh, y);  // 586 blocks

    dim3 rgrid(C_OUT, 8);
    reduce_partial_kernel<<<rgrid, 256, 0, stream>>>(y, partial);
    finalize_stats_kernel<<<1, 64, 0, stream>>>(partial, stats);

    dim3 bgrid((N_NODES + 255) / 256, C_OUT);
    bn_relu_kernel<<<bgrid, 256, 0, stream>>>(y, stats, gamma, beta, out);
}
